// FreeNoiseCogVideoXBlock_56968446214852
// MI455X (gfx1250) — matrix operand order repaired and checked by reading, not compiled, not run on hardware
//
#include <hip/hip_runtime.h>
#include <math.h>

// ---------------- problem constants ----------------
#define NF      24
#define S_      144
#define T_      128
#define DIM     1024
#define HEADS   16
#define HD      64
#define TEMB    512
#define CL      8
#define STRIDE_ 4
#define W_      5           // windows: starts = {0,4,8,12,16}
#define LV      (CL * S_)   // 1152
#define L_      (T_ + LV)   // 1280
#define NROW    (NF * S_)   // 3456
#define MX      (W_ * L_)   // 6400  (rows of x / q / k / v / o)
#define FFM     (T_ + NROW) // 3584
#define FFH     4096

typedef __bf16 bf16;
typedef __bf16 bf16x8 __attribute__((ext_vector_type(8)));
typedef __bf16 v16bf  __attribute__((ext_vector_type(16)));
typedef float  v8f    __attribute__((ext_vector_type(8)));

// ---------------- TDM availability probe ----------------
#if defined(__gfx1250__) && __has_builtin(__builtin_amdgcn_tensor_load_to_lds) && \
    __has_builtin(__builtin_amdgcn_s_wait_tensorcnt)
#define HAVE_TDM 1
#else
#define HAVE_TDM 0
#endif

#if HAVE_TDM
typedef unsigned int u32x4 __attribute__((ext_vector_type(4)));
typedef int          i32x8 __attribute__((ext_vector_type(8)));
typedef int          i32x4 __attribute__((ext_vector_type(4)));

// 2D tensor tile load via Tensor Data Mover (D# per ISA cdna5 ch.8).
// All sizes/strides in elements (data_size = 2 bytes, bf16).
// 6-arg builtin form (amdgpu-toolchain): (g0, g1, g2, g3, g4, cpol)
__device__ __forceinline__ void tdm_load_2d(unsigned lds_addr, const void* gptr,
                                            unsigned tensor_d0, unsigned tensor_d1,
                                            unsigned long long stride0,
                                            unsigned tile_d0, unsigned tile_d1) {
    unsigned long long ga = (unsigned long long)(size_t)gptr;
    u32x4 g0;
    g0[0] = 1u;                                    // count=1, user mode
    g0[1] = lds_addr;                              // lds_addr (bytes)
    g0[2] = (unsigned)(ga & 0xFFFFFFFFu);          // global_addr[31:0]
    g0[3] = (unsigned)((ga >> 32) & 0x01FFFFFFu) | (2u << 30);   // addr[56:32] | type=2
    i32x8 g1;
    g1[0] = 0x00010000;                            // data_size=1 (2 bytes)
    g1[1] = (int)((tensor_d0 & 0xFFFFu) << 16);    // tensor_dim0 [15:0]
    g1[2] = (int)(((tensor_d0 >> 16) & 0xFFFFu) | ((tensor_d1 & 0xFFFFu) << 16));
    g1[3] = (int)(((tensor_d1 >> 16) & 0xFFFFu) | ((tile_d0 & 0xFFFFu) << 16));
    g1[4] = (int)(tile_d1 & 0xFFFFu);              // tile_dim1; tile_dim2=0
    g1[5] = (int)(unsigned)(stride0 & 0xFFFFFFFFull);
    g1[6] = (int)(unsigned)((stride0 >> 32) & 0xFFFFull);  // dim1_stride=0
    g1[7] = 0;
    i32x4 z4; z4[0] = 0; z4[1] = 0; z4[2] = 0; z4[3] = 0;
    i32x8 z8;
#pragma unroll
    for (int i = 0; i < 8; ++i) z8[i] = 0;
    __builtin_amdgcn_tensor_load_to_lds(g0, g1, z4, z4, z8, 0);
}
#endif

// ---------------- helpers ----------------
__device__ __forceinline__ bf16 f2b(float f) { return (bf16)f; }

__device__ __forceinline__ v8f zero8() {
    v8f z;
#pragma unroll
    for (int i = 0; i < 8; ++i) z[i] = 0.0f;
    return z;
}

union FragU { v16bf v; bf16x8 h[2]; };

// A-fragment (16-bit A 16x32, ISA 7.12.2): lane m=lane&15, half=lane>>4
// elements 0..7  = K[half*8 .. half*8+7]
// elements 8..15 = K[16+half*8 .. 16+half*8+7]
__device__ __forceinline__ v16bf ldA(const bf16* base, int lda) {
    int lane = threadIdx.x & 31;
    int m = lane & 15, hf = lane >> 4;
    const bf16* p = base + m * lda;
    FragU u;
    u.h[0] = *(const bf16x8*)(p + hf * 8);
    u.h[1] = *(const bf16x8*)(p + 16 + hf * 8);
    return u.v;
}

// B-fragment (16-bit B 32x16): lane col n=lane&15, half=lane>>4;
// elements 0..7 = K[half*8..], elements 8..15 = K[16+half*8..].  baseT is a [N][ldb] column-major-K tile.
__device__ __forceinline__ v16bf ldB(const bf16* baseT, int ldb) {
    int lane = threadIdx.x & 31;
    int n = lane & 15, hf = lane >> 4;
    const bf16* p = baseT + n * ldb;
    FragU u;
    u.h[0] = *(const bf16x8*)(p + hf * 8);
    u.h[1] = *(const bf16x8*)(p + 16 + hf * 8);
    return u.v;
}

__device__ __forceinline__ v8f wmma_bf(v16bf a, v16bf b, v8f c) {
    return __builtin_amdgcn_wmma_f32_16x16x32_bf16(false, a, false, b, (short)0, c,
                                                   false, false);
}

__device__ __forceinline__ float rmax16(float v) {
#pragma unroll
    for (int o = 8; o; o >>= 1) v = fmaxf(v, __shfl_xor(v, o, 32));
    return v;
}
__device__ __forceinline__ float rsum16(float v) {
#pragma unroll
    for (int o = 8; o; o >>= 1) v += __shfl_xor(v, o, 32);
    return v;
}
__device__ __forceinline__ float rsum32(float v) {
#pragma unroll
    for (int o = 16; o; o >>= 1) v += __shfl_xor(v, o, 32);
    return v;
}

__device__ __forceinline__ float gelu_tanh(float x) {
    return 0.5f * x * (1.0f + tanhf(0.7978845608028654f * (x + 0.044715f * x * x * x)));
}

// ---------------- f32 -> bf16 convert ----------------
__global__ __launch_bounds__(256) void cvt_kernel(const float* __restrict__ s,
                                                  bf16* __restrict__ d, int n) {
    int i = blockIdx.x * 256 + threadIdx.x;
    if (i < n) d[i] = f2b(s[i]);
}

// ---------------- emb = silu(temb) @ W + b  (W: TEMB x 6*DIM) ----------------
__global__ __launch_bounds__(256) void emb_kernel(const float* __restrict__ temb,
                                                  const float* __restrict__ w,
                                                  const float* __restrict__ b,
                                                  float* __restrict__ out) {
    __shared__ float st[TEMB];
    int tid = threadIdx.x;
    for (int i = tid; i < TEMB; i += 256) {
        float t = temb[i];
        st[i] = t / (1.0f + __expf(-t));
    }
    __syncthreads();
    int j = blockIdx.x * 256 + tid;
    float a = b[j];
    for (int i = 0; i < TEMB; ++i) a += st[i] * w[(size_t)i * (6 * DIM) + j];
    out[j] = a;
}

// ---------------- build x (LN + adaLN modulate), bf16 out ----------------
__global__ __launch_bounds__(256) void build_x_kernel(const float* __restrict__ hs,
                                                      const float* __restrict__ ehs,
                                                      const float* __restrict__ emb,
                                                      const float* __restrict__ g,
                                                      const float* __restrict__ bt,
                                                      bf16* __restrict__ xb) {
    __shared__ float r1[256], r2[256];
    int row = blockIdx.x;
    int w = row / L_, l = row % L_;
    const float *src, *sc, *sh;
    if (l < T_) { src = ehs + (size_t)l * DIM;  sc = emb + 4 * DIM; sh = emb + 3 * DIM; }
    else {
        int p = l - T_;
        src = hs + (size_t)(w * STRIDE_ * S_ + p) * DIM;
        sc = emb + DIM; sh = emb;
    }
    int tid = threadIdx.x;
    float x[4], s = 0.f, ss = 0.f;
#pragma unroll
    for (int c = 0; c < 4; ++c) {
        x[c] = src[tid + c * 256];
        s += x[c]; ss += x[c] * x[c];
    }
    r1[tid] = s; r2[tid] = ss; __syncthreads();
    for (int o = 128; o; o >>= 1) {
        if (tid < o) { r1[tid] += r1[tid + o]; r2[tid] += r2[tid + o]; }
        __syncthreads();
    }
    float mean = r1[0] * (1.0f / DIM);
    float var = r2[0] * (1.0f / DIM) - mean * mean;
    float rs = rsqrtf(var + 1e-5f);
#pragma unroll
    for (int c = 0; c < 4; ++c) {
        int col = tid + c * 256;
        float v = (x[c] - mean) * rs * g[col] + bt[col];
        v = v * (1.0f + sc[col]) + sh[col];
        xb[(size_t)row * DIM + col] = f2b(v);
    }
}

// ---------------- generic bf16 WMMA GEMM: C = A(MxK) @ B(KxN) [+bias][gelu] ----
// block tile 128x64, BK=32; 256 thr = 8 waves, wave tile 32x32 (2x2 wmma)
// A tile staged via Tensor Data Mover (one wave issues, TENSORcnt-waited);
// B tile staged per-thread with transpose-on-store.
__global__ __launch_bounds__(256) void gemm_bf16_kernel(const bf16* __restrict__ A,
                                                        const bf16* __restrict__ B,
                                                        float* __restrict__ Cf,
                                                        bf16* __restrict__ Cb,
                                                        const float* __restrict__ bias,
                                                        int M, int N, int K, int act) {
    __shared__ bf16 sA[128 * 32];
    __shared__ bf16 sBt[64 * 32];
    int tid = threadIdx.x, lane = tid & 31, wid = tid >> 5;
    int wm = wid >> 1, wn = wid & 1;
    int rowBlk = blockIdx.y * 128, colBlk = blockIdx.x * 64;

#if HAVE_TDM
    unsigned sA_lds = (unsigned)(size_t)(void*)sA;
#endif

    v8f acc[2][2];
#pragma unroll
    for (int i = 0; i < 2; ++i)
#pragma unroll
        for (int j = 0; j < 2; ++j) acc[i][j] = zero8();

    for (int k0 = 0; k0 < K; k0 += 32) {
#if HAVE_TDM
        // A tile (128 rows x 32 K) via TDM: tensor dims (K, M), stride K
        if (wid == 0) {
            tdm_load_2d(sA_lds, A + (size_t)rowBlk * K + k0,
                        (unsigned)K, (unsigned)M, (unsigned long long)K, 32u, 128u);
        }
#else
#pragma unroll
        for (int r = 0; r < 2; ++r) {
            int idx = tid + r * 256;
            int ar = idx >> 2, ac = (idx & 3) * 8;
            *(bf16x8*)(sA + ar * 32 + ac) =
                *(const bf16x8*)(A + (size_t)(rowBlk + ar) * K + k0 + ac);
        }
#endif
        // stage B tile transposed -> sBt[N=64][K=32]
        {
            int bk = tid >> 3, bn = (tid & 7) * 8;
            bf16x8 bv = *(const bf16x8*)(B + (size_t)(k0 + bk) * N + colBlk + bn);
#pragma unroll
            for (int j = 0; j < 8; ++j) sBt[(bn + j) * 32 + bk] = bv[j];
        }
        if (k0 + 32 < K) {
            __builtin_prefetch(B + (size_t)(k0 + 32 + (tid >> 3)) * N + colBlk, 0, 0);
        }
#if HAVE_TDM
        if (wid == 0) __builtin_amdgcn_s_wait_tensorcnt(0);
#endif
        __syncthreads();

        v16bf a0 = ldA(sA + (wm * 32) * 32, 32);
        v16bf a1 = ldA(sA + (wm * 32 + 16) * 32, 32);
        v16bf b0 = ldB(sBt + (wn * 32) * 32, 32);
        v16bf b1 = ldB(sBt + (wn * 32 + 16) * 32, 32);
        acc[0][0] = wmma_bf(a0, b0, acc[0][0]);
        acc[0][1] = wmma_bf(a0, b1, acc[0][1]);
        acc[1][0] = wmma_bf(a1, b0, acc[1][0]);
        acc[1][1] = wmma_bf(a1, b1, acc[1][1]);
        __syncthreads();
    }

    int hf = lane >> 4, nl = lane & 15;
#pragma unroll
    for (int i = 0; i < 2; ++i)
#pragma unroll
        for (int j = 0; j < 2; ++j)
#pragma unroll
            for (int r = 0; r < 8; ++r) {
                int m = wm * 32 + i * 16 + r + 8 * hf;
                int n = wn * 32 + j * 16 + nl;
                size_t o = (size_t)(rowBlk + m) * N + (colBlk + n);
                float v = acc[i][j][r];
                if (bias) v += bias[colBlk + n];
                if (act) v = gelu_tanh(v);
                if (Cf) Cf[o] = v; else Cb[o] = f2b(v);
            }
}

// ---------------- per-head LN (QK-norm) + repack to [w][h][l][hd] bf16 --------
__global__ __launch_bounds__(256) void headln_kernel(const float* __restrict__ src,
                                                     bf16* __restrict__ dst,
                                                     const float* __restrict__ g,
                                                     const float* __restrict__ b,
                                                     int do_ln) {
    int pair = blockIdx.x * 8 + (threadIdx.x >> 5);  // (row, head)
    int lane = threadIdx.x & 31;
    int row = pair >> 4, h = pair & 15;
    int w = row / L_, l = row % L_;
    const float* p = src + (size_t)row * DIM + h * HD;
    float v0 = p[lane * 2], v1 = p[lane * 2 + 1];
    if (do_ln) {
        float mean = rsum32(v0 + v1) * (1.0f / HD);
        float d0 = v0 - mean, d1 = v1 - mean;
        float var = rsum32(d0 * d0 + d1 * d1) * (1.0f / HD);
        float rs = rsqrtf(var + 1e-6f);
        v0 = d0 * rs * g[lane * 2] + b[lane * 2];
        v1 = d1 * rs * g[lane * 2 + 1] + b[lane * 2 + 1];
    }
    bf16* q = dst + ((size_t)(w * HEADS + h) * L_ + l) * HD;
    q[lane * 2] = f2b(v0);
    q[lane * 2 + 1] = f2b(v1);
}

// ---------------- flash attention: per block 64 q-rows of one (w,h) ----------
// grid (L_/64, HEADS, W_), block 128 (4 waves, 16 q-rows each)
// K tile staged via TDM; V tile staged per-thread (needs transpose).
__global__ __launch_bounds__(128) void attn_kernel(const bf16* __restrict__ Qh,
                                                   const bf16* __restrict__ Kh,
                                                   const bf16* __restrict__ Vh,
                                                   bf16* __restrict__ Ob) {
    __shared__ bf16 sK[64 * 64];   // [key][hd]
    __shared__ bf16 sVt[64 * 64];  // [hd][key]
    __shared__ bf16 sP[4][16 * 64];

    int tid = threadIdx.x, lane = tid & 31, wvid = tid >> 5;
    int hf = lane >> 4, nl = lane & 15;
    int q0 = blockIdx.x * 64, h = blockIdx.y, w = blockIdx.z;
    size_t headOff = (size_t)(w * HEADS + h) * L_ * HD;

#if HAVE_TDM
    unsigned sK_lds = (unsigned)(size_t)(void*)sK;
#endif

    // resident Q fragments (K-dim = HD = 64 -> 2 fragments)
    const bf16* qp = Qh + headOff + (size_t)(q0 + wvid * 16 + nl) * HD;
    FragU qa0, qa1;
    qa0.h[0] = *(const bf16x8*)(qp + hf * 8);
    qa0.h[1] = *(const bf16x8*)(qp + 16 + hf * 8);
    qa1.h[0] = *(const bf16x8*)(qp + 32 + hf * 8);
    qa1.h[1] = *(const bf16x8*)(qp + 48 + hf * 8);

    float mrow[8], lrow[8];
    v8f oacc[4];
#pragma unroll
    for (int r = 0; r < 8; ++r) { mrow[r] = -3.0e38f; lrow[r] = 0.0f; }
#pragma unroll
    for (int j = 0; j < 4; ++j) oacc[j] = zero8();

    for (int kt = 0; kt < L_ / 64; ++kt) {
        const bf16* Kb = Kh + headOff + (size_t)(kt * 64) * HD;
        const bf16* Vb = Vh + headOff + (size_t)(kt * 64) * HD;
#if HAVE_TDM
        if (wvid == 0) {
            // 64 keys x 64 hd straight copy into sK via TDM
            tdm_load_2d(sK_lds, Kb, (unsigned)HD, (unsigned)L_,
                        (unsigned long long)HD, 64u, 64u);
        }
#pragma unroll
        for (int r = 0; r < 4; ++r) {
            int idx = tid + r * 128;
            int krow = idx >> 3, c8 = (idx & 7) * 8;
            bf16x8 vv = *(const bf16x8*)(Vb + krow * 64 + c8);
#pragma unroll
            for (int j = 0; j < 8; ++j) sVt[(c8 + j) * 64 + krow] = vv[j];
        }
        if (wvid == 0) __builtin_amdgcn_s_wait_tensorcnt(0);
#else
#pragma unroll
        for (int r = 0; r < 4; ++r) {
            int idx = tid + r * 128;
            int krow = idx >> 3, c8 = (idx & 7) * 8;
            *(bf16x8*)(sK + krow * 64 + c8) = *(const bf16x8*)(Kb + krow * 64 + c8);
            bf16x8 vv = *(const bf16x8*)(Vb + krow * 64 + c8);
#pragma unroll
            for (int j = 0; j < 8; ++j) sVt[(c8 + j) * 64 + krow] = vv[j];
        }
#endif
        __syncthreads();

        // S = Q @ K^T  (K-dim = hd)
        v8f sc[4];
#pragma unroll
        for (int j = 0; j < 4; ++j) sc[j] = zero8();
#pragma unroll
        for (int s = 0; s < 2; ++s) {
            v16bf aa = s ? qa1.v : qa0.v;
#pragma unroll
            for (int j = 0; j < 4; ++j) {
                v16bf bb = ldB(sK + (j * 16) * 64 + s * 32, 64);
                sc[j] = wmma_bf(aa, bb, sc[j]);
            }
        }
        // scale + online softmax (rows live as (r, lane-half))
#pragma unroll
        for (int j = 0; j < 4; ++j)
#pragma unroll
            for (int r = 0; r < 8; ++r) sc[j][r] *= 0.125f;
#pragma unroll
        for (int r = 0; r < 8; ++r) {
            float mx = -3.0e38f;
#pragma unroll
            for (int j = 0; j < 4; ++j) mx = fmaxf(mx, sc[j][r]);
            mx = rmax16(mx);
            float newm = fmaxf(mrow[r], mx);
            float fold = __expf(mrow[r] - newm);
            float rs = 0.0f;
#pragma unroll
            for (int j = 0; j < 4; ++j) {
                float p = __expf(sc[j][r] - newm);
                sc[j][r] = p;
                rs += p;
            }
            rs = rsum16(rs);
            lrow[r] = lrow[r] * fold + rs;
#pragma unroll
            for (int j = 0; j < 4; ++j) oacc[j][r] *= fold;
            mrow[r] = newm;
        }
        // P -> LDS (C-layout -> memory), then reload as A fragments
#pragma unroll
        for (int j = 0; j < 4; ++j)
#pragma unroll
            for (int r = 0; r < 8; ++r)
                sP[wvid][(r + 8 * hf) * 64 + j * 16 + nl] = f2b(sc[j][r]);
        __syncthreads();

        // O += P @ V  (K-dim = keys)
#pragma unroll
        for (int s = 0; s < 2; ++s) {
            v16bf pa = ldA(sP[wvid] + s * 32, 64);
#pragma unroll
            for (int j = 0; j < 4; ++j) {
                v16bf vb = ldB(sVt + (j * 16) * 64 + s * 32, 64);
                oacc[j] = wmma_bf(pa, vb, oacc[j]);
            }
        }
        __syncthreads();
    }

    // normalize + write o as [w*L + q][h*64 + d] bf16
#pragma unroll
    for (int j = 0; j < 4; ++j)
#pragma unroll
        for (int r = 0; r < 8; ++r) {
            float v = oacc[j][r] / lrow[r];
            size_t o = (size_t)(w * L_ + q0 + wvid * 16 + r + 8 * hf) * DIM +
                       h * HD + j * 16 + nl;
            Ob[o] = f2b(v);
        }
}

// ---------------- pyramid-weight window blend + encoder mean -----------------
__global__ __launch_bounds__(256) void combine_kernel(const float* __restrict__ hs,
                                                      const float* __restrict__ ehs,
                                                      const float* __restrict__ attn,
                                                      const float* __restrict__ emb,
                                                      float* __restrict__ outH,
                                                      float* __restrict__ outE) {
    const float pyr[8] = {1.f, 2.f, 3.f, 4.f, 4.f, 3.f, 2.f, 1.f};
    int row = blockIdx.x, tid = threadIdx.x;
    if (row < NROW) {
        int f = row / S_, s = row % S_;
        const float* gate = emb + 2 * DIM;
#pragma unroll
        for (int c = 0; c < 4; ++c) {
            int col = tid + c * 256;
            float acc = 0.f, cnt = 0.f;
            float hv = hs[(size_t)row * DIM + col];
            float gv = gate[col];
            for (int w = 0; w < W_; ++w) {
                int off = f - w * STRIDE_;
                if (off >= 0 && off < CL) {
                    float wt = pyr[off];
                    cnt += wt;
                    size_t arow = (size_t)(w * L_ + T_ + off * S_ + s);
                    acc += wt * (hv + gv * attn[arow * DIM + col]);
                }
            }
            outH[(size_t)row * DIM + col] = acc / cnt;
        }
    } else {
        int t = row - NROW;
        const float* egate = emb + 5 * DIM;
#pragma unroll
        for (int c = 0; c < 4; ++c) {
            int col = tid + c * 256;
            float sm = 0.f;
            for (int w = 0; w < W_; ++w)
                sm += attn[(size_t)(w * L_ + t) * DIM + col];
            outE[(size_t)t * DIM + col] =
                ehs[(size_t)t * DIM + col] + egate[col] * (sm * (1.0f / W_));
        }
    }
}

// ---------------- build ff_in (LN + modulate of [enc; hidden]) ----------------
__global__ __launch_bounds__(256) void build_ffin_kernel(const float* __restrict__ hid,
                                                         const float* __restrict__ enc,
                                                         const float* __restrict__ emb,
                                                         const float* __restrict__ g,
                                                         const float* __restrict__ bt,
                                                         bf16* __restrict__ xb) {
    __shared__ float r1[256], r2[256];
    int row = blockIdx.x;
    const float *src, *sc, *sh;
    if (row < T_) { src = enc + (size_t)row * DIM; sc = emb + 4 * DIM; sh = emb + 3 * DIM; }
    else          { src = hid + (size_t)(row - T_) * DIM; sc = emb + DIM; sh = emb; }
    int tid = threadIdx.x;
    float x[4], s = 0.f, ss = 0.f;
#pragma unroll
    for (int c = 0; c < 4; ++c) {
        x[c] = src[tid + c * 256];
        s += x[c]; ss += x[c] * x[c];
    }
    r1[tid] = s; r2[tid] = ss; __syncthreads();
    for (int o = 128; o; o >>= 1) {
        if (tid < o) { r1[tid] += r1[tid + o]; r2[tid] += r2[tid + o]; }
        __syncthreads();
    }
    float mean = r1[0] * (1.0f / DIM);
    float var = r2[0] * (1.0f / DIM) - mean * mean;
    float rs = rsqrtf(var + 1e-5f);
#pragma unroll
    for (int c = 0; c < 4; ++c) {
        int col = tid + c * 256;
        float v = (x[c] - mean) * rs * g[col] + bt[col];
        v = v * (1.0f + sc[col]) + sh[col];
        xb[(size_t)row * DIM + col] = f2b(v);
    }
}

// ---------------- final gated residual (in place on d_out) --------------------
__global__ __launch_bounds__(256) void final_kernel(const float* __restrict__ ffo,
                                                    const float* __restrict__ emb2,
                                                    float* __restrict__ outH,
                                                    float* __restrict__ outE) {
    int row = blockIdx.x, tid = threadIdx.x;
#pragma unroll
    for (int c = 0; c < 4; ++c) {
        int col = tid + c * 256;
        float f = ffo[(size_t)row * DIM + col];
        if (row < T_)
            outE[(size_t)row * DIM + col] += (emb2 + 5 * DIM)[col] * f;
        else
            outH[(size_t)(row - T_) * DIM + col] += (emb2 + 2 * DIM)[col] * f;
    }
}

// ---------------- host-side orchestration ----------------
extern "C" void kernel_launch(void* const* d_in, const int* in_sizes, int n_in,
                              void* d_out, int out_size, void* d_ws, size_t ws_size,
                              hipStream_t stream) {
    (void)in_sizes; (void)n_in; (void)out_size; (void)ws_size;
    const float* hs   = (const float*)d_in[0];
    const float* ehs  = (const float*)d_in[1];
    const float* temb = (const float*)d_in[2];
    const float* n1w  = (const float*)d_in[3];
    const float* n1bv = (const float*)d_in[4];
    const float* n1g  = (const float*)d_in[5];
    const float* n1b  = (const float*)d_in[6];
    const float* wq   = (const float*)d_in[7];
    const float* wk   = (const float*)d_in[8];
    const float* wv   = (const float*)d_in[9];
    const float* nqg  = (const float*)d_in[10];
    const float* nqb  = (const float*)d_in[11];
    const float* nkg  = (const float*)d_in[12];
    const float* nkb  = (const float*)d_in[13];
    const float* wo   = (const float*)d_in[14];
    const float* bo   = (const float*)d_in[15];
    const float* n2w  = (const float*)d_in[16];
    const float* n2bv = (const float*)d_in[17];
    const float* n2g  = (const float*)d_in[18];
    const float* n2b  = (const float*)d_in[19];
    const float* fw1  = (const float*)d_in[20];
    const float* fb1  = (const float*)d_in[21];
    const float* fw2  = (const float*)d_in[22];
    const float* fb2  = (const float*)d_in[23];

    char* ws = (char*)d_ws;
    size_t off = 0;
    auto alloc = [&](size_t bytes) -> void* {
        void* p = ws + off;
        off = (off + bytes + 255) & ~(size_t)255;
        return p;
    };

    const size_t szW  = (size_t)DIM * DIM * sizeof(bf16);
    const size_t szF1 = (size_t)DIM * FFH * sizeof(bf16);
    const size_t szXB = (size_t)MX * DIM * sizeof(bf16);
    const size_t szQF = (size_t)MX * DIM * sizeof(float);

    bf16*  wqB  = (bf16*)alloc(szW);
    bf16*  wkB  = (bf16*)alloc(szW);
    bf16*  wvB  = (bf16*)alloc(szW);
    bf16*  woB  = (bf16*)alloc(szW);
    bf16*  w1B  = (bf16*)alloc(szF1);
    bf16*  w2B  = (bf16*)alloc(szF1);
    float* emb1 = (float*)alloc(6 * DIM * sizeof(float));
    float* emb2 = (float*)alloc(6 * DIM * sizeof(float));
    bf16*  xb   = (bf16*)alloc(szXB);          // reused as o_bf after QKV
    bf16*  qh   = (bf16*)alloc(szXB);
    bf16*  kh   = (bf16*)alloc(szXB);
    bf16*  vh   = (bf16*)alloc(szXB);
    char*  big  = (char*)alloc(3 * szQF);      // phase1: Q/K/V f32; phase2: attn/ff

    // phase-1 views
    float* Qf = (float*)big;
    float* Kf = (float*)(big + szQF);
    float* Vf = (float*)(big + 2 * szQF);
    // phase-2 views (lifetimes disjoint from Q/K/V f32)
    float* attnF = (float*)big;
    bf16*  ffin  = (bf16*)(big + szQF);
    bf16*  ffh   = (bf16*)(big + szQF + (size_t)FFM * DIM * sizeof(bf16));
    float* ffo   = (float*)(big + szQF + (size_t)FFM * DIM * sizeof(bf16) +
                            (size_t)FFM * FFH * sizeof(bf16));
    bf16*  obf   = xb;

    float* outH = (float*)d_out;
    float* outE = outH + (size_t)NROW * DIM;

    // 1) weights -> bf16
    cvt_kernel<<<(DIM * DIM) / 256, 256, 0, stream>>>(wq, wqB, DIM * DIM);
    cvt_kernel<<<(DIM * DIM) / 256, 256, 0, stream>>>(wk, wkB, DIM * DIM);
    cvt_kernel<<<(DIM * DIM) / 256, 256, 0, stream>>>(wv, wvB, DIM * DIM);
    cvt_kernel<<<(DIM * DIM) / 256, 256, 0, stream>>>(wo, woB, DIM * DIM);
    cvt_kernel<<<(DIM * FFH) / 256, 256, 0, stream>>>(fw1, w1B, DIM * FFH);
    cvt_kernel<<<(DIM * FFH) / 256, 256, 0, stream>>>(fw2, w2B, DIM * FFH);

    // 2) adaLN embeddings
    emb_kernel<<<(6 * DIM) / 256, 256, 0, stream>>>(temb, n1w, n1bv, emb1);
    emb_kernel<<<(6 * DIM) / 256, 256, 0, stream>>>(temb, n2w, n2bv, emb2);

    // 3) build normalized x (bf16)
    build_x_kernel<<<MX, 256, 0, stream>>>(hs, ehs, emb1, n1g, n1b, xb);

    // 4) QKV projections
    dim3 gQKV(DIM / 64, MX / 128);
    gemm_bf16_kernel<<<gQKV, 256, 0, stream>>>(xb, wqB, Qf, nullptr, nullptr,
                                               MX, DIM, DIM, 0);
    gemm_bf16_kernel<<<gQKV, 256, 0, stream>>>(xb, wkB, Kf, nullptr, nullptr,
                                               MX, DIM, DIM, 0);
    gemm_bf16_kernel<<<gQKV, 256, 0, stream>>>(xb, wvB, Vf, nullptr, nullptr,
                                               MX, DIM, DIM, 0);

    // 5) QK-norm + head repack (V just repacked)
    int lnBlocks = (MX * HEADS) / 8;
    headln_kernel<<<lnBlocks, 256, 0, stream>>>(Qf, qh, nqg, nqb, 1);
    headln_kernel<<<lnBlocks, 256, 0, stream>>>(Kf, kh, nkg, nkb, 1);
    headln_kernel<<<lnBlocks, 256, 0, stream>>>(Vf, vh, nqg, nqb, 0);

    // 6) flash attention  (writes o_bf, reusing xb storage)
    dim3 gAttn(L_ / 64, HEADS, W_);
    attn_kernel<<<gAttn, 128, 0, stream>>>(qh, kh, vh, obf);

    // 7) output projection
    gemm_bf16_kernel<<<gQKV, 256, 0, stream>>>(obf, woB, attnF, nullptr, bo,
                                               MX, DIM, DIM, 0);

    // 8) pyramid blend -> intermediate hidden/enc staged in d_out
    combine_kernel<<<NROW + T_, 256, 0, stream>>>(hs, ehs, attnF, emb1, outH, outE);

    // 9) FFN
    build_ffin_kernel<<<FFM, 256, 0, stream>>>(outH, outE, emb2, n2g, n2b, ffin);
    dim3 gFF1(FFH / 64, FFM / 128);
    gemm_bf16_kernel<<<gFF1, 256, 0, stream>>>(ffin, w1B, nullptr, ffh, fb1,
                                               FFM, FFH, DIM, 1);
    dim3 gFF2(DIM / 64, FFM / 128);
    gemm_bf16_kernel<<<gFF2, 256, 0, stream>>>(ffh, w2B, ffo, nullptr, fb2,
                                               FFM, DIM, FFH, 0);

    // 10) final gated residual (in place on d_out)
    final_kernel<<<FFM, 256, 0, stream>>>(ffo, emb2, outH, outE);
}
